// GRUD_68762426409572
// MI455X (gfx1250) — matrix operand order repaired and checked by reading, not compiled, not run on hardware
//
#include <hip/hip_runtime.h>
#include <hip/hip_bf16.h>

typedef __attribute__((ext_vector_type(16))) __bf16 v16bf;
typedef __attribute__((ext_vector_type(8)))  float  v8f;

#define Bsz 256
#define Tsz 256
#define Fsz 128
#define Hsz 256
#define Ksz 512                 // F + H + F
#define QN  (Bsz * Tsz)         // 65536 flattened (t,b) rows
#define Z8  {0.f,0.f,0.f,0.f,0.f,0.f,0.f,0.f}

// ---- packed weight layouts (ushort units) ----
// Wxm : xm-segment B-fragments, 48 N-tiles (3 gates x 16), 8 K-chunks
#define WXM_USH   (48 * 8 * 512)            // 196608
// Whs : h-segment B-fragments, 3 gates x 16 N-tiles x 8 K-chunks
#define WHS_USH   (3 * 16 * 8 * 512)        // 196608
// Wgh : delta_h B-fragments, 16 N-tiles x 4 K-chunks
#define WGH_USH   (16 * 4 * 512)            // 32768
#define PACK_USH  (WXM_USH + WHS_USH + WGH_USH)   // 425984

// ---- workspace byte offsets (requires ~186 MB of d_ws) ----
#define OFF_WXM  ((size_t)0)
#define OFF_WHS  ((size_t)WXM_USH * 2)                    // 393216
#define OFF_WGH  (OFF_WHS + (size_t)WHS_USH * 2)          // 786432
#define OFF_AXM  (OFF_WGH + (size_t)WGH_USH * 2)          // 851968   : bf16 [QN][256]
#define OFF_AD   (OFF_AXM + (size_t)QN * 256 * 2)         // +33.6MB  : bf16 [QN][128]
#define OFF_PRE  (OFF_AD  + (size_t)QN * 128 * 2)         // +16.8MB  : bf16 [QN][768] ([t][b][gate*256+n])
#define OFF_DH   (OFF_PRE + (size_t)QN * 768 * 2)         // +100.7MB : bf16 [QN][256]
#define OFF_O    (OFF_DH  + (size_t)QN * 256 * 2)         // +33.6MB  : f32 [256][2]

__device__ __forceinline__ unsigned short f2bf(float f) {
    unsigned int x = __float_as_uint(f);
    return (unsigned short)((x + 0x7FFFu + ((x >> 16) & 1u)) >> 16);
}
__device__ __forceinline__ float bf2f(unsigned short u) {
    return __uint_as_float(((unsigned int)u) << 16);
}

// A fragment (16x32 bf16, row-major in LDS), contiguous k order (k = 16*hi + s)
__device__ __forceinline__ v16bf load_a_frag(const unsigned short* A, int astride, int kc) {
    int lane = threadIdx.x & 31;
    int row  = lane & 15;
    int hi   = (lane >> 4) & 1;
    const unsigned short* p = A + row * astride + kc * 32 + hi * 16;
    union { v16bf v; unsigned int u[8]; } f;
#pragma unroll
    for (int q = 0; q < 8; ++q) {
        int kk = 2 * q;
        f.u[q] = *(const unsigned int*)(p + kk);
    }
    return f.v;
}
// B fragment: 16 contiguous ushorts per lane (32B aligned)
__device__ __forceinline__ v16bf load_b_frag(const unsigned short* Bp) {
    return *(const v16bf*)(Bp + (threadIdx.x & 31) * 16);
}
__device__ __forceinline__ v8f wmma_bf16(v16bf a, v16bf b, v8f c) {
    return __builtin_amdgcn_wmma_f32_16x16x32_bf16(false, a, false, b, (short)0, c, false, false);
}
// async bulk copy global -> LDS (16B), tracked by ASYNCcnt
__device__ __forceinline__ void async_g2l_16(unsigned lds_off, const void* g) {
    asm volatile("global_load_async_to_lds_b128 %0, %1, off" :: "v"(lds_off), "v"(g) : "memory");
}

// =============== pack: f32 weights -> bf16 WMMA-B fragments ===============
__global__ void grud_pack(const float* __restrict__ Wz, const float* __restrict__ Wr,
                          const float* __restrict__ Wh, const float* __restrict__ Wgh,
                          unsigned short* __restrict__ ws) {
    int idx = blockIdx.x * blockDim.x + threadIdx.x;
    if (idx >= PACK_USH) return;
    unsigned short v;
    if (idx < WXM_USH) {                      // xm-segments, N-tiles 0..47
        int i = idx & 15, lane = (idx >> 4) & 31, kc = (idx >> 9) & 7, nT = idx >> 12;
        int g = nT >> 4;
        int col = ((nT & 15) << 4) + (lane & 15);
        int K = kc * 32 + ((lane >> 4) & 1) * 16 + i;        // 0..255
        int ksrc = (K < 128) ? K : (K + 256);                // x cols 0..127, m cols 384..511
        const float* W = (g == 0) ? Wz : (g == 1) ? Wr : Wh;
        v = f2bf(W[col * Ksz + ksrc]);
    } else if (idx < WXM_USH + WHS_USH) {     // h-segments
        int r = idx - WXM_USH;
        int i = r & 15, lane = (r >> 4) & 31, kc = (r >> 9) & 7, nT = (r >> 12) & 15, g = r >> 16;
        int col = (nT << 4) + (lane & 15);
        int ksrc = 128 + kc * 32 + ((lane >> 4) & 1) * 16 + i;   // h cols 128..383
        const float* W = (g == 0) ? Wz : (g == 1) ? Wr : Wh;
        v = f2bf(W[col * Ksz + ksrc]);
    } else {                                  // Wgh
        int r = idx - WXM_USH - WHS_USH;
        int i = r & 15, lane = (r >> 4) & 31, kc = (r >> 9) & 3, nT = r >> 11;
        int col = (nT << 4) + (lane & 15);
        int K = kc * 32 + ((lane >> 4) & 1) * 16 + i;
        v = f2bf(Wgh[col * Fsz + K]);
    }
    ws[idx] = v;
}

// =============== A1: imputation -> Axm bf16 [q][x|m], Ad bf16 ===============
__global__ void grud_impute(const float* __restrict__ X,  const float* __restrict__ Xl,
                            const float* __restrict__ Mk, const float* __restrict__ Dl,
                            const float* __restrict__ Xm, const float* __restrict__ Wgx,
                            const float* __restrict__ bgx,
                            unsigned short* __restrict__ Axm, unsigned short* __restrict__ Ad) {
    int idx = blockIdx.x * blockDim.x + threadIdx.x;     // q*128 + f, q = t*B + b
    if (idx >= QN * Fsz) return;
    int f = idx & 127, q = idx >> 7, b = q & 255, t = q >> 8;
    size_t s = ((size_t)b * Tsz + t) * Fsz + f;
    float d = Dl[s], m = Mk[s], xv = X[s], xlv = Xl[s], xmv = Xm[t * Fsz + f];
    float dlx = __expf(-fmaxf(0.f, d * Wgx[f * Fsz + f] + bgx[f]));
    float x = m * xv + (1.f - m) * (dlx * xlv + (1.f - dlx) * xmv);
    Axm[q * 256 + f]       = f2bf(x);
    Axm[q * 256 + 128 + f] = f2bf(m);
    Ad[q * 128 + f]        = f2bf(d);
}

// =============== A2: pre = [x,m]@Wxm^T + bias  (QN x 768) ===============
__global__ void __launch_bounds__(512)
grud_pregemm(const unsigned short* __restrict__ Axm, const unsigned short* __restrict__ Wxm,
             const float* __restrict__ bz, const float* __restrict__ br,
             const float* __restrict__ bh, unsigned short* __restrict__ pre) {
    __shared__ unsigned short at[16 * 256];
    const int tid = threadIdx.x, wave = tid >> 5, lane = tid & 31;
    const int n = lane & 15, hi = (lane >> 4) & 1;
    const int q0 = blockIdx.x * 16;
    const int nT = blockIdx.y * 16 + wave;          // 0..47
    // stage A tile (16 rows x 256 bf16 = 8 KB), coalesced
    {
        const uint4* src = (const uint4*)(Axm + (size_t)q0 * 256);
        uint4* dst = (uint4*)at;
        dst[tid] = src[tid];
        dst[tid + 512] = src[tid + 512];
    }
    __syncthreads();
    v8f acc = Z8;
#pragma unroll
    for (int kc = 0; kc < 8; ++kc) {
        v16bf a = load_a_frag(at, 256, kc);
        v16bf b = load_b_frag(Wxm + ((size_t)nT * 8 + kc) * 512);
        acc = wmma_bf16(a, b, acc);
    }
    const int g = nT >> 4;
    const float* bias = (g == 0) ? bz : (g == 1) ? br : bh;
    float bc = bias[((nT & 15) << 4) + n];
#pragma unroll
    for (int p = 0; p < 8; ++p) {
        int row = q0 + p + 8 * hi;
        pre[(size_t)row * 768 + nT * 16 + n] = f2bf(acc[p] + bc);
    }
}

// =============== A3: dh = exp(-relu(d@Wgh^T + bgh))  (QN x 256) ===============
__global__ void __launch_bounds__(512)
grud_dhgemm(const unsigned short* __restrict__ Ad, const unsigned short* __restrict__ Wgh,
            const float* __restrict__ bgh, unsigned short* __restrict__ dh) {
    __shared__ unsigned short at[16 * 128];
    const int tid = threadIdx.x, wave = tid >> 5, lane = tid & 31;
    const int n = lane & 15, hi = (lane >> 4) & 1;
    const int q0 = blockIdx.x * 16;
    if (tid < 256) ((uint4*)at)[tid] = ((const uint4*)(Ad + (size_t)q0 * 128))[tid];
    __syncthreads();
    v8f acc = Z8;
#pragma unroll
    for (int kc = 0; kc < 4; ++kc) {
        v16bf a = load_a_frag(at, 128, kc);
        v16bf b = load_b_frag(Wgh + ((size_t)wave * 4 + kc) * 512);
        acc = wmma_bf16(a, b, acc);
    }
    float bc = bgh[wave * 16 + n];
#pragma unroll
    for (int p = 0; p < 8; ++p) {
        int row = q0 + p + 8 * hi;
        dh[(size_t)row * 256 + wave * 16 + n] = f2bf(__expf(-fmaxf(0.f, acc[p] + bc)));
    }
}

// =============== B: recurrence, 16 persistent WGs (one B-tile each) ===============
// LDS (dynamic 278528B): Wz_h 128KB | Wr_h 128KB | h bf16 8KB | r*h bf16 8KB
__global__ void __launch_bounds__(512)
grud_recur(const unsigned short* __restrict__ Whs,   // packed h-seg fragments (g0,g1,g2)
           const unsigned short* __restrict__ pre,   // [t][b][768] bf16
           const unsigned short* __restrict__ dh,    // [t][b][256] bf16
           const float* __restrict__ Wfc, const float* __restrict__ bfc,
           float* __restrict__ o_all) {
    extern __shared__ unsigned short smem[];
    unsigned short* wz_lds = smem;               // 65536 ush
    unsigned short* wr_lds = smem + 65536;       // 65536 ush
    unsigned short* habuf  = smem + 131072;      // 4096 ush : h bf16 [16][256]
    unsigned short* rhbuf  = smem + 135168;      // 4096 ush : r*h bf16 [16][256]

    const int tid = threadIdx.x, wave = tid >> 5, lane = tid & 31;
    const int n = lane & 15, hi = (lane >> 4) & 1;
    const int col = wave * 16 + n;
    const int b0 = blockIdx.x * 16;

    // stage Wz_h + Wr_h (262144 B) into LDS with async copies
    {
        unsigned base = (unsigned)(unsigned long long)(void*)smem;
        const char* g = (const char*)Whs;
        for (int j = tid; j < 262144 / 16; j += 512)
            async_g2l_16(base + j * 16, g + j * 16);
        asm volatile("s_wait_asynccnt 0" ::: "memory");
    }
    // W_h h-segment fragments resident in registers (64 VGPRs/lane)
    v16bf whf[8];
#pragma unroll
    for (int kc = 0; kc < 8; ++kc)
        whf[kc] = load_b_frag(Whs + ((size_t)(32 + wave) * 8 + kc) * 512);
    __syncthreads();

    v8f hreg = Z8;                               // h slab in registers
    const int lb_pre = (b0 + 8 * hi) * 768 + col;   // lane base into pre rows
    const int lb_dh  = (b0 + 8 * hi) * 256 + col;

    for (int t = 0; t < Tsz; ++t) {
        const size_t tp = (size_t)t * Bsz * 768;
        const size_t td = (size_t)t * Bsz * 256;
        // phase 1: h *= delta_h[t]; publish h bf16 to LDS
#pragma unroll
        for (int p = 0; p < 8; ++p) {
            int M = p + 8 * hi;
            hreg[p] *= bf2f(dh[td + lb_dh + p * 256]);
            habuf[M * 256 + col] = f2bf(hreg[p]);
        }
        if (t + 1 < Tsz) {
            __builtin_prefetch(dh + td + Bsz * 256 + lb_dh, 0, 3);
            __builtin_prefetch(pre + tp + Bsz * 768 + lb_pre, 0, 3);
            __builtin_prefetch(pre + tp + Bsz * 768 + lb_pre + 512, 0, 3);
        }
        __syncthreads();

        // phase 2: z, r from LDS-resident weights; write r*h
        v8f zacc = Z8, racc = Z8;
#pragma unroll
        for (int kc = 0; kc < 8; ++kc) {
            v16bf a  = load_a_frag(habuf, 256, kc);
            v16bf bz = load_b_frag(wz_lds + (wave * 8 + kc) * 512);
            v16bf br = load_b_frag(wr_lds + (wave * 8 + kc) * 512);
            zacc = wmma_bf16(a, bz, zacc);
            racc = wmma_bf16(a, br, racc);
        }
        v8f zreg;
#pragma unroll
        for (int p = 0; p < 8; ++p) {
            int M = p + 8 * hi;
            float z = 1.f / (1.f + __expf(-(zacc[p] + bf2f(pre[tp + lb_pre + p * 768]))));
            float r = 1.f / (1.f + __expf(-(racc[p] + bf2f(pre[tp + lb_pre + p * 768 + 256]))));
            zreg[p] = z;
            rhbuf[M * 256 + col] = f2bf(r * hreg[p]);
        }
        __syncthreads();

        // phase 3: h_tilde from register-resident W_h; GRU update
        v8f hacc = Z8;
#pragma unroll
        for (int kc = 0; kc < 8; ++kc) {
            v16bf a = load_a_frag(rhbuf, 256, kc);
            hacc = wmma_bf16(a, whf[kc], hacc);
        }
#pragma unroll
        for (int p = 0; p < 8; ++p) {
            float ht = tanhf(hacc[p] + bf2f(pre[tp + lb_pre + p * 768 + 512]));
            hreg[p] = (1.f - zreg[p]) * hreg[p] + zreg[p] * ht;
        }
        __syncthreads();
    }

    // epilogue: o = hT @ Wfc^T + bfc (stash f32 h in re-used LDS)
    float* hfin = (float*)smem;
#pragma unroll
    for (int p = 0; p < 8; ++p) hfin[(p + 8 * hi) * 256 + col] = hreg[p];
    __syncthreads();
    if (tid < 32) {
        int row = tid >> 1, j = tid & 1;
        float s = bfc[j];
        for (int k = 0; k < Hsz; ++k) s += hfin[row * 256 + k] * Wfc[j * Hsz + k];
        o_all[(b0 + row) * 2 + j] = s;
    }
}

// =============== BN (training-mode batch stats) ===============
__global__ void grud_bn(const float* __restrict__ o, const float* __restrict__ gma,
                        const float* __restrict__ bta, float* __restrict__ out) {
    __shared__ float s0[256], s1[256], q0[256], q1[256];
    int t = threadIdx.x;
    float a = o[t * 2 + 0], b = o[t * 2 + 1];
    s0[t] = a; s1[t] = b; q0[t] = a * a; q1[t] = b * b;
    __syncthreads();
    for (int s = 128; s > 0; s >>= 1) {
        if (t < s) { s0[t] += s0[t + s]; s1[t] += s1[t + s]; q0[t] += q0[t + s]; q1[t] += q1[t + s]; }
        __syncthreads();
    }
    float mu0 = s0[0] * (1.f / 256.f), mu1 = s1[0] * (1.f / 256.f);
    float v0 = q0[0] * (1.f / 256.f) - mu0 * mu0, v1 = q1[0] * (1.f / 256.f) - mu1 * mu1;
    out[t * 2 + 0] = (a - mu0) * rsqrtf(v0 + 1e-5f) * gma[0] + bta[0];
    out[t * 2 + 1] = (b - mu1) * rsqrtf(v1 + 1e-5f) * gma[1] + bta[1];
}

extern "C" void kernel_launch(void* const* d_in, const int* in_sizes, int n_in,
                              void* d_out, int out_size, void* d_ws, size_t ws_size,
                              hipStream_t stream) {
    const float* X   = (const float*)d_in[0];
    const float* Xl  = (const float*)d_in[1];
    const float* Mk  = (const float*)d_in[2];
    const float* Dl  = (const float*)d_in[3];
    const float* Xm  = (const float*)d_in[4];
    const float* Wgx = (const float*)d_in[5];
    const float* bgx = (const float*)d_in[6];
    const float* Wgh = (const float*)d_in[7];
    const float* bgh = (const float*)d_in[8];
    const float* Wz  = (const float*)d_in[9];
    const float* bz  = (const float*)d_in[10];
    const float* Wr  = (const float*)d_in[11];
    const float* br  = (const float*)d_in[12];
    const float* Wh  = (const float*)d_in[13];
    const float* bh  = (const float*)d_in[14];
    const float* Wfc = (const float*)d_in[15];
    const float* bfc = (const float*)d_in[16];
    const float* gma = (const float*)d_in[17];
    const float* bta = (const float*)d_in[18];

    char* ws = (char*)d_ws;
    unsigned short* pWxm = (unsigned short*)(ws + OFF_WXM);
    unsigned short* pWhs = (unsigned short*)(ws + OFF_WHS);
    unsigned short* pWgh = (unsigned short*)(ws + OFF_WGH);
    unsigned short* Axm  = (unsigned short*)(ws + OFF_AXM);
    unsigned short* Ad   = (unsigned short*)(ws + OFF_AD);
    unsigned short* pre  = (unsigned short*)(ws + OFF_PRE);
    unsigned short* dhb  = (unsigned short*)(ws + OFF_DH);
    float* o_all = (float*)(ws + OFF_O);

    grud_pack<<<(PACK_USH + 255) / 256, 256, 0, stream>>>(Wz, Wr, Wh, Wgh, pWxm);
    grud_impute<<<(QN * Fsz) / 256, 256, 0, stream>>>(X, Xl, Mk, Dl, Xm, Wgx, bgx, Axm, Ad);
    grud_pregemm<<<dim3(QN / 16, 3), 512, 0, stream>>>(Axm, pWxm, bz, br, bh, pre);
    grud_dhgemm<<<QN / 16, 512, 0, stream>>>(Ad, pWgh, bgh, dhb);
    grud_recur<<<Bsz / 16, 512, 278528, stream>>>(pWhs, pre, dhb, Wfc, bfc, o_all);
    grud_bn<<<1, 256, 0, stream>>>(o_all, gma, bta, (float*)d_out);
}
